// EquivariantMessagePassing_30262339568375
// MI455X (gfx1250) — compile-verified, matrix operand order checked by reading, not run on hardware
//
#include <hip/hip_runtime.h>
#include <hip/hip_bf16.h>

typedef __attribute__((ext_vector_type(16))) _Float16 v16h;
typedef __attribute__((ext_vector_type(8)))  float    v8f;

#define N_NODES 20000
#define N_EDGES 320000
#define D_IN    160          // 64 scalar + 32*3 vector
#define MUL_S   64
#define MUL_V   32

#define INV_SQ3 0.57735026918962576f

// fragment-layout k index for 16-bit 16x32 A/B tiles:
// lane half-select hs = lane>>4; element e (0..15): VGPR j=e/2 holds 2 halfs.
// j<4 -> K = hs*8 + j*2 + (e&1); j>=4 -> K = 16 + hs*8 + (j-4)*2 + (e&1)
__device__ __forceinline__ int kloc(int hs, int e) {
    int j = e >> 1, rem = e & 1;
    return (j < 4) ? (hs * 8 + j * 2 + rem) : (16 + hs * 8 + (j - 4) * 2 + rem);
}

__device__ __forceinline__ void wait_asynccnt0() {
#if defined(__has_builtin)
#if __has_builtin(__builtin_amdgcn_s_wait_asynccnt)
    __builtin_amdgcn_s_wait_asynccnt(0);
    return;
#endif
#endif
    asm volatile("s_wait_asynccnt 0" ::: "memory");
}

// ---------------------------------------------------------------- zero d_out
__global__ void zero_out_kernel(float* __restrict__ out) {
    int i = blockIdx.x * 256 + threadIdx.x;
    if (i < N_NODES * D_IN) out[i] = 0.0f;
}

// ----------------------------------------------- fuse W0@V0, W1@V1 into f16
// U0 frags: [kb(3)][nt(4)][lane(32)][e(16)] halfs   (6144 halfs)
// U1 frags: [kb(3)][nt(2)][lane(32)][e(16)] halfs   (3072 halfs) at offset 6144
__global__ void precompute_kernel(const float* __restrict__ W0,
                                  const float* __restrict__ W1,
                                  const float* __restrict__ V0,
                                  const float* __restrict__ V1,
                                  _Float16* __restrict__ U) {
    const float sc0 = 1.0f / (sqrtf(96.0f) * sqrtf(128.0f));
    const float sc1 = 1.0f / (sqrtf(96.0f) * sqrtf(64.0f));
    int tid = threadIdx.x;
    // U0 = W0(96x128) @ V0(128x64), scaled, fragment layout
    for (int fidx = tid; fidx < 6144; fidx += 256) {
        int e    = fidx & 15;
        int lane = (fidx >> 4) & 31;
        int nt   = (fidx >> 9) & 3;
        int kb   = fidx >> 11;
        int n = nt * 16 + (lane & 15);
        int k = kb * 32 + kloc(lane >> 4, e);
        float acc = 0.0f;
        for (int h = 0; h < 128; ++h) acc += W0[k * 128 + h] * V0[h * 64 + n];
        U[fidx] = (_Float16)(acc * sc0);
    }
    // U1 = W1(96x64) @ V1(64x32), scaled, fragment layout
    for (int fidx = tid; fidx < 3072; fidx += 256) {
        int e    = fidx & 15;
        int lane = (fidx >> 4) & 31;
        int nt   = (fidx >> 9) & 1;
        int kb   = fidx >> 10;
        int n = nt * 16 + (lane & 15);
        int k = kb * 32 + kloc(lane >> 4, e);
        float acc = 0.0f;
        for (int h = 0; h < 64; ++h) acc += W1[k * 64 + h] * V1[h * 32 + n];
        U[6144 + fidx] = (_Float16)(acc * sc1);
    }
}

// -------------------------------------------------------------- main kernel
// 128 threads = 4 waves; each wave owns a 16-edge tile.
// Staging: per-lane GLOBAL_LOAD_ASYNC_TO_LDS_B128 fills the 16x160 float tile
// directly into LDS (20 chunks/lane), waited on before any B-fragment load.
// Compute: o0 (16x64) via 3x4 WMMA f32_16x16x32_f16, o1 (16x32 x3 comps)
// via 3x(3x2) WMMAs, scatter-add into out[dst] with global f32 atomics.
__global__ void __launch_bounds__(128)
edge_msg_kernel(const float* __restrict__ feat,
                const int*   __restrict__ eidx,
                const float* __restrict__ eattr,
                const _Float16* __restrict__ U,
                float* __restrict__ out) {
    __shared__ float sfeat[4][16][164]; // 160 features + 4 edge_attr; row = 656B (16B mult)
    __shared__ int   sdst[4][16];

    const int wv   = threadIdx.x >> 5;
    const int lane = threadIdx.x & 31;
    const int base = (blockIdx.x * 4 + wv) * 16;

    // ------- stage: async-gather 16 edges' node features into LDS -------
    int srcIdx = 0;
    if (lane < 16) {
        srcIdx = eidx[base + lane];
        sdst[wv][lane] = eidx[N_EDGES + base + lane];
        const float* ea = eattr + (size_t)(base + lane) * 4;
        sfeat[wv][lane][160] = ea[0];
        sfeat[wv][lane][161] = ea[1];
        sfeat[wv][lane][162] = ea[2];
        sfeat[wv][lane][163] = ea[3];
    }
    // 16 rows x 640B = 40 x 16B chunks per row; 20 iterations of 32 lanes
    #pragma unroll 4
    for (int it = 0; it < 20; ++it) {
        int idx   = it * 32 + lane;        // 0..639
        int row   = idx / 40;              // edge row in tile
        int chunk = idx % 40;              // 16B chunk within row
        int s     = __shfl(srcIdx, row);   // divergent-row gather of src index
        const float* gp = feat + (size_t)s * D_IN + chunk * 4;
        uint64_t gaddr  = (uint64_t)(uintptr_t)gp;
        uint32_t laddr  = (uint32_t)(uintptr_t)&sfeat[wv][row][chunk * 4];
        asm volatile("global_load_async_to_lds_b128 %0, %1, off"
                     :: "v"(laddr), "v"(gaddr) : "memory");
    }
    wait_asynccnt0();
    __syncthreads();

    const int m  = lane & 15;   // edge row inside tile (A/C layout)
    const int hs = lane >> 4;   // K half-select for A fragments
    const float a0  = sfeat[wv][m][160];
    const float a1x = sfeat[wv][m][161];
    const float a1y = sfeat[wv][m][162];
    const float a1z = sfeat[wv][m][163];
    const float* fr = &sfeat[wv][m][0];

    const v16h* Bf = (const v16h*)U;   // U0 frags at v16h idx 0, U1 at 384

    // ------------------------------ o0 = t0 @ U0 ------------------------------
    {
        v8f acc[4] = {v8f{}, v8f{}, v8f{}, v8f{}};
        #pragma unroll
        for (int kb = 0; kb < 3; ++kb) {
            v16h a;
            #pragma unroll
            for (int e = 0; e < 16; ++e) {
                int k = kb * 32 + kloc(hs, e);
                float v;
                if (k < MUL_S) {
                    v = fr[k] * a0;                          // A = s * e0
                } else {
                    int u = (k - MUL_S) * 3 + MUL_S;         // D = (v . e1)/sqrt3
                    v = (fr[u] * a1x + fr[u + 1] * a1y + fr[u + 2] * a1z) * INV_SQ3;
                }
                a[e] = (_Float16)v;
            }
            #pragma unroll
            for (int nt = 0; nt < 4; ++nt) {
                v16h b = Bf[(kb * 4 + nt) * 32 + lane];
                acc[nt] = __builtin_amdgcn_wmma_f32_16x16x32_f16(
                    false, a, false, b, (short)0, acc[nt], false, false);
            }
        }
        // scatter o0: feature f = nt*16 + (lane&15); row = r + 8*hs
        #pragma unroll
        for (int nt = 0; nt < 4; ++nt) {
            int f = nt * 16 + (lane & 15);
            #pragma unroll
            for (int r = 0; r < 8; ++r) {
                int row = r + 8 * hs;
                atomicAdd(&out[(size_t)sdst[wv][row] * D_IN + f], acc[nt][r]);
            }
        }
    }

    // --------------------------- o1[i] = t1o[:,:,i] @ U1 ----------------------
    #pragma unroll
    for (int comp = 0; comp < 3; ++comp) {
        const float e1c = (comp == 0) ? a1x : (comp == 1) ? a1y : a1z;
        v8f acc[2] = {v8f{}, v8f{}};
        #pragma unroll
        for (int kb = 0; kb < 3; ++kb) {
            v16h a;
            #pragma unroll
            for (int e = 0; e < 16; ++e) {
                int k = kb * 32 + kloc(hs, e);
                float v;
                if (k < MUL_S) {
                    v = fr[k] * e1c;                          // B = s * e1_i
                } else {
                    v = fr[(k - MUL_S) * 3 + MUL_S + comp] * a0;  // C = v_i * e0
                }
                a[e] = (_Float16)v;
            }
            #pragma unroll
            for (int nt = 0; nt < 2; ++nt) {
                v16h b = Bf[384 + (kb * 2 + nt) * 32 + lane];
                acc[nt] = __builtin_amdgcn_wmma_f32_16x16x32_f16(
                    false, a, false, b, (short)0, acc[nt], false, false);
            }
        }
        // scatter o1: u = nt*16 + (lane&15), feature f = 64 + u*3 + comp
        #pragma unroll
        for (int nt = 0; nt < 2; ++nt) {
            int u = nt * 16 + (lane & 15);
            int f = MUL_S + u * 3 + comp;
            #pragma unroll
            for (int r = 0; r < 8; ++r) {
                int row = r + 8 * hs;
                atomicAdd(&out[(size_t)sdst[wv][row] * D_IN + f], acc[nt][r]);
            }
        }
    }
}

// ---------------------------------------------------------- layernorm (in place)
__global__ void __launch_bounds__(128)
layernorm_kernel(float* __restrict__ out,
                 const float* __restrict__ gamma,
                 const float* __restrict__ beta) {
    const int wv   = threadIdx.x >> 5;
    const int lane = threadIdx.x & 31;
    const int node = blockIdx.x * 4 + wv;
    if (node >= N_NODES) return;

    float vals[5];
    float sum = 0.0f, sq = 0.0f;
    #pragma unroll
    for (int c = 0; c < 5; ++c) {
        float v = out[(size_t)node * D_IN + c * 32 + lane];
        vals[c] = v;
        sum += v;
        sq  += v * v;
    }
    #pragma unroll
    for (int off = 16; off > 0; off >>= 1) {
        sum += __shfl_xor(sum, off, 32);
        sq  += __shfl_xor(sq,  off, 32);
    }
    const float mu  = sum * (1.0f / D_IN);
    const float var = sq * (1.0f / D_IN) - mu * mu;
    const float inv = rsqrtf(var + 1e-5f);
    #pragma unroll
    for (int c = 0; c < 5; ++c) {
        int f = c * 32 + lane;
        out[(size_t)node * D_IN + f] = (vals[c] - mu) * inv * gamma[f] + beta[f];
    }
}

extern "C" void kernel_launch(void* const* d_in, const int* in_sizes, int n_in,
                              void* d_out, int out_size, void* d_ws, size_t ws_size,
                              hipStream_t stream) {
    const float* node_features = (const float*)d_in[0];
    const int*   edge_index    = (const int*)d_in[1];
    const float* edge_attr     = (const float*)d_in[2];
    // d_in[3] node_pos: unused (dead in reference after DCE)
    const float* W0 = (const float*)d_in[4];
    const float* W1 = (const float*)d_in[5];
    // W2, W3 (d_in[6], d_in[7]) feed h1e/h2e which are deleted -> dead
    const float* V0 = (const float*)d_in[8];
    const float* V1 = (const float*)d_in[9];
    const float* gamma = (const float*)d_in[10];
    const float* beta  = (const float*)d_in[11];

    float* out = (float*)d_out;
    _Float16* U = (_Float16*)d_ws;   // 9216 halfs = 18 KB of fused-weight fragments

    zero_out_kernel<<<(N_NODES * D_IN + 255) / 256, 256, 0, stream>>>(out);
    precompute_kernel<<<1, 256, 0, stream>>>(W0, W1, V0, V1, U);
    edge_msg_kernel<<<N_EDGES / 64, 128, 0, stream>>>(
        node_features, edge_index, edge_attr, U, out);
    layernorm_kernel<<<(N_NODES + 3) / 4, 128, 0, stream>>>(out, gamma, beta);
}
